// SpatialAttentionConvolution_40243843564211
// MI455X (gfx1250) — compile-verified, matrix operand order checked by reading, not run on hardware
//
#include <hip/hip_runtime.h>

// Problem constants (match reference)
#define BB 4
#define CC 32
#define HH 2048
#define WW 512
#define PLANE (HH * WW)          // 1<<20

typedef __attribute__((ext_vector_type(16))) _Float16 v16h;
typedef __attribute__((ext_vector_type(8)))  _Float16 v8h;
typedef __attribute__((ext_vector_type(8)))  float    v8f;

// ---------------------------------------------------------------------------
// Kernel 1: fused 1x1 convs C->1 for q,k,v.  x:(B,C,H,W) f32 -> q/k/v:(B,H,W) f16
// ---------------------------------------------------------------------------
__global__ void k_qkv(const float* __restrict__ x,
                      const float* __restrict__ Wq, const float* __restrict__ bq,
                      const float* __restrict__ Wk, const float* __restrict__ bk,
                      const float* __restrict__ Wv, const float* __restrict__ bv,
                      _Float16* __restrict__ qh, _Float16* __restrict__ kh,
                      _Float16* __restrict__ vh) {
    __shared__ float sWq[CC], sWk[CC], sWv[CC];
    int tid = threadIdx.x;
    if (tid < CC) { sWq[tid] = Wq[tid]; sWk[tid] = Wk[tid]; sWv[tid] = Wv[tid]; }
    __syncthreads();
    size_t t  = (size_t)blockIdx.x * 256 + tid;          // < B*H*W
    int    b  = (int)(t >> 20);                          // PLANE == 2^20
    size_t hw = t & (size_t)(PLANE - 1);
    const float* xp = x + (size_t)b * CC * PLANE + hw;
    float q = 0.f, k = 0.f, v = 0.f;
#pragma unroll
    for (int c = 0; c < CC; ++c) {
        float xv = xp[(size_t)c * PLANE];
        q += xv * sWq[c];
        k += xv * sWk[c];
        v += xv * sWv[c];
    }
    q += bq[0]; k += bk[0]; v += bv[0];
    qh[t] = (_Float16)q; kh[t] = (_Float16)k; vh[t] = (_Float16)v;
}

// ---------------------------------------------------------------------------
// Kernel 2: transpose v:(B,H,W) -> vT:(B,W,H), f16, 32x32 LDS tiles
// ---------------------------------------------------------------------------
__global__ void k_transpose_v(const _Float16* __restrict__ vh,
                              _Float16* __restrict__ vT) {
    __shared__ _Float16 tile[32][33];
    int b  = blockIdx.z;
    int w0 = blockIdx.x * 32;
    int h0 = blockIdx.y * 32;
    const _Float16* src = vh + (size_t)b * PLANE;
    _Float16*       dst = vT + (size_t)b * PLANE;
#pragma unroll
    for (int j = 0; j < 4; ++j) {
        int hh = threadIdx.y + j * 8;
        tile[hh][threadIdx.x] = src[(size_t)(h0 + hh) * WW + w0 + threadIdx.x];
    }
    __syncthreads();
#pragma unroll
    for (int j = 0; j < 4; ++j) {
        int ww = threadIdx.y + j * 8;
        dst[(size_t)(w0 + ww) * HH + h0 + threadIdx.x] = tile[threadIdx.x][ww];
    }
}

// ---------------------------------------------------------------------------
// WMMA 16x32 f16 tile loader (A layout; B uses the mirrored lane=N layout,
// which is identical when the operand is stored with its N index row-major).
//  lane 0-15 : row = lane,    K chunks [kb+0..7]  and [kb+16..23]
//  lane 16-31: row = lane-16, K chunks [kb+8..15] and [kb+24..31]
// ---------------------------------------------------------------------------
__device__ __forceinline__ v16h load_tile16x32(const _Float16* __restrict__ base,
                                               int row0, int ld, int kb) {
    int lane = threadIdx.x & 31;
    int m    = lane & 15;
    int hi   = lane >> 4;
    const _Float16* p = base + (size_t)(row0 + m) * ld + kb + hi * 8;
    union { v16h v; v8h h[2]; } u;
    u.h[0] = *(const v8h*)(p);        // global_load_b128
    u.h[1] = *(const v8h*)(p + 16);   // global_load_b128
    return u.v;
}

__device__ __forceinline__ v8f wmma_f16(v16h a, v16h b, v8f c) {
    return __builtin_amdgcn_wmma_f32_16x16x32_f16(false, a, false, b,
                                                  (short)0, c, false, false);
}

// ---------------------------------------------------------------------------
// Kernel 3: S = Q * K^T  per batch.  M=N=H, K=W.
// Block = 8 waves (256 thr) -> 128x64 block tile; wave -> 32x32 (2x2 WMMA).
// ---------------------------------------------------------------------------
__global__ void k_gemm_qk(const _Float16* __restrict__ qh,
                          const _Float16* __restrict__ kh,
                          float* __restrict__ S) {
    int b    = blockIdx.z;
    int wave = threadIdx.x >> 5;
    int m0   = blockIdx.x * 128 + (wave >> 1) * 32;
    int n0   = blockIdx.y * 64  + (wave & 1) * 32;
    const _Float16* A  = qh + (size_t)b * PLANE;
    const _Float16* Bm = kh + (size_t)b * PLANE;
    v8f acc[2][2] = {};
    for (int kb = 0; kb < WW; kb += 32) {
        v16h a0 = load_tile16x32(A,  m0,      WW, kb);
        v16h a1 = load_tile16x32(A,  m0 + 16, WW, kb);
        v16h b0 = load_tile16x32(Bm, n0,      WW, kb);
        v16h b1 = load_tile16x32(Bm, n0 + 16, WW, kb);
        acc[0][0] = wmma_f16(a0, b0, acc[0][0]);
        acc[0][1] = wmma_f16(a0, b1, acc[0][1]);
        acc[1][0] = wmma_f16(a1, b0, acc[1][0]);
        acc[1][1] = wmma_f16(a1, b1, acc[1][1]);
    }
    int lane = threadIdx.x & 31;
    int nl   = lane & 15;
    int hi   = lane >> 4;
#pragma unroll
    for (int i = 0; i < 2; ++i)
#pragma unroll
        for (int j = 0; j < 2; ++j) {
            float* out = S + ((size_t)b * HH + m0 + i * 16 + hi * 8) * HH
                           + n0 + j * 16 + nl;
#pragma unroll
            for (int r = 0; r < 8; ++r)
                out[(size_t)r * HH] = acc[i][j][r];
        }
}

// ---------------------------------------------------------------------------
// Kernel 4: row softmax over S (length H), writes f16 probs IN PLACE over the
// f32 row (first half of the row's bytes).  One block (256 thr) per row.
// ---------------------------------------------------------------------------
__global__ void k_softmax(float* __restrict__ S) {
    __shared__ float red[256];
    float* row = S + (size_t)blockIdx.x * HH;
    int t = threadIdx.x;
    float vals[8];
    float m = -3.4e38f;
#pragma unroll
    for (int j = 0; j < 8; ++j) {
        vals[j] = row[t + j * 256];
        m = fmaxf(m, vals[j]);
    }
    red[t] = m;
    __syncthreads();
    for (int s = 128; s > 0; s >>= 1) {
        if (t < s) red[t] = fmaxf(red[t], red[t + s]);
        __syncthreads();
    }
    float rmax = red[0];
    __syncthreads();
    float sum = 0.f;
#pragma unroll
    for (int j = 0; j < 8; ++j) {
        vals[j] = __expf(vals[j] - rmax);
        sum += vals[j];
    }
    red[t] = sum;
    __syncthreads();
    for (int s = 128; s > 0; s >>= 1) {
        if (t < s) red[t] += red[t + s];
        __syncthreads();
    }
    float inv = 1.0f / red[0];
    __syncthreads();                       // all f32 reads done before f16 writes
    _Float16* P = (_Float16*)row;
#pragma unroll
    for (int j = 0; j < 8; ++j)
        P[t + j * 256] = (_Float16)(vals[j] * inv);
}

// ---------------------------------------------------------------------------
// Kernel 5: O = P * V   (P f16 packed in S rows, ld = 2*H halves; B from vT)
// M=H, N=W, K=H.  Same 128x64 block / 32x32 wave tiling.
// ---------------------------------------------------------------------------
__global__ void k_gemm_pv(const float* __restrict__ S,
                          const _Float16* __restrict__ vT,
                          float* __restrict__ O) {
    int b    = blockIdx.z;
    int wave = threadIdx.x >> 5;
    int m0   = blockIdx.x * 128 + (wave >> 1) * 32;
    int n0   = blockIdx.y * 64  + (wave & 1) * 32;
    const _Float16* A  = (const _Float16*)(S + (size_t)b * HH * HH); // ld = 2*H halves
    const _Float16* Bm = vT + (size_t)b * PLANE;                     // ld = H
    v8f acc[2][2] = {};
    for (int kb = 0; kb < HH; kb += 32) {
        v16h a0 = load_tile16x32(A,  m0,      2 * HH, kb);
        v16h a1 = load_tile16x32(A,  m0 + 16, 2 * HH, kb);
        v16h b0 = load_tile16x32(Bm, n0,      HH,     kb);
        v16h b1 = load_tile16x32(Bm, n0 + 16, HH,     kb);
        acc[0][0] = wmma_f16(a0, b0, acc[0][0]);
        acc[0][1] = wmma_f16(a0, b1, acc[0][1]);
        acc[1][0] = wmma_f16(a1, b0, acc[1][0]);
        acc[1][1] = wmma_f16(a1, b1, acc[1][1]);
    }
    int lane = threadIdx.x & 31;
    int nl   = lane & 15;
    int hi   = lane >> 4;
#pragma unroll
    for (int i = 0; i < 2; ++i)
#pragma unroll
        for (int j = 0; j < 2; ++j) {
            float* out = O + ((size_t)b * HH + m0 + i * 16 + hi * 8) * WW
                           + n0 + j * 16 + nl;
#pragma unroll
            for (int r = 0; r < 8; ++r)
                out[(size_t)r * WW] = acc[i][j][r];
        }
}

// ---------------------------------------------------------------------------
// Kernel 6: z[b,c,h,w] = O[b,h,w]*W2[c] + b2[c]
// ---------------------------------------------------------------------------
__global__ void k_expand(const float* __restrict__ O,
                         const float* __restrict__ W2,
                         const float* __restrict__ b2,
                         float* __restrict__ z) {
    __shared__ float sW[CC], sB[CC];
    int tid = threadIdx.x;
    if (tid < CC) { sW[tid] = W2[tid]; sB[tid] = b2[tid]; }
    __syncthreads();
    size_t t  = (size_t)blockIdx.x * 256 + tid;
    int    b  = (int)(t >> 20);
    size_t hw = t & (size_t)(PLANE - 1);
    float  o  = O[t];
    float* zp = z + (size_t)b * CC * PLANE + hw;
#pragma unroll
    for (int c = 0; c < CC; ++c)
        zp[(size_t)c * PLANE] = o * sW[c] + sB[c];
}

// ---------------------------------------------------------------------------
// Host launcher
// ---------------------------------------------------------------------------
extern "C" void kernel_launch(void* const* d_in, const int* in_sizes, int n_in,
                              void* d_out, int out_size, void* d_ws, size_t ws_size,
                              hipStream_t stream) {
    const float* x  = (const float*)d_in[0];
    const float* Wq = (const float*)d_in[1];
    const float* bq = (const float*)d_in[2];
    const float* Wk = (const float*)d_in[3];
    const float* bk = (const float*)d_in[4];
    const float* Wv = (const float*)d_in[5];
    const float* bv = (const float*)d_in[6];
    const float* W2 = (const float*)d_in[7];
    const float* b2 = (const float*)d_in[8];
    float* z = (float*)d_out;

    // Workspace carve-up (total ~112 MB)
    char* ws = (char*)d_ws;
    const size_t SZ_HALF = (size_t)BB * PLANE * sizeof(_Float16);  // 8 MB
    _Float16* qh = (_Float16*)(ws);
    _Float16* kh = (_Float16*)(ws + SZ_HALF);
    _Float16* vh = (_Float16*)(ws + 2 * SZ_HALF);
    _Float16* vT = (_Float16*)(ws + 3 * SZ_HALF);
    float*    S  = (float*)(ws + 4 * SZ_HALF);                               // 67 MB
    float*    O  = (float*)(ws + 4 * SZ_HALF + (size_t)BB * HH * HH * 4);    // 16 MB

    const int nBHW = BB * PLANE;  // 4,194,304

    k_qkv<<<nBHW / 256, 256, 0, stream>>>(x, Wq, bq, Wk, bk, Wv, bv, qh, kh, vh);

    k_transpose_v<<<dim3(WW / 32, HH / 32, BB), dim3(32, 8), 0, stream>>>(vh, vT);

    k_gemm_qk<<<dim3(HH / 128, HH / 64, BB), 256, 0, stream>>>(qh, kh, S);

    k_softmax<<<BB * HH, 256, 0, stream>>>(S);

    k_gemm_pv<<<dim3(HH / 128, WW / 64, BB), 256, 0, stream>>>(S, vT, O);

    k_expand<<<nBHW / 256, 256, 0, stream>>>(O, W2, b2, z);
}
